// Denoiser_78151224918570
// MI455X (gfx1250) — matrix operand order repaired and checked by reading, not compiled, not run on hardware
//
#include <hip/hip_runtime.h>
#include <hip/hip_bf16.h>

// ---------------------------------------------------------------------------
// Types / helpers
// ---------------------------------------------------------------------------
typedef __attribute__((ext_vector_type(16))) __bf16 v16bf;
typedef __attribute__((ext_vector_type(8)))  float  v8f;

#define NEG_SLOPE 0.01f

static __device__ __forceinline__ __bf16 f2bf(float f) {
  // round-to-nearest-even fp32 -> bf16 (used only in the one-time weight cvt)
  unsigned u = __float_as_uint(f);
  unsigned r = (u + 0x7FFFu + ((u >> 16) & 1u)) >> 16;
  unsigned short s = (unsigned short)r;
  __bf16 b;
  __builtin_memcpy(&b, &s, sizeof(b));
  return b;
}

static __device__ __forceinline__ float apply_act(float v, int act) {
  if (act == 1) return fmaxf(v, 0.0f);
  if (act == 2) return v > 0.0f ? v : NEG_SLOPE * v;
  return v;
}

// one-time fp32 -> bf16 weight conversion
__global__ void k_cvt_bf16(const float* __restrict__ in,
                           __bf16* __restrict__ out, long n) {
  long i = (long)blockIdx.x * blockDim.x + threadIdx.x;
  if (i < n) out[i] = f2bf(in[i]);
}

// ---------------------------------------------------------------------------
// Implicit-GEMM 3x3 SAME conv via V_WMMA_F32_16X16X32_BF16, tap-major:
//   out = sum_t  W_t[Cout x Cin] x shift_t(x)[Cin x pixels]
// One wave -> 16 out-channels x 32 pixels (two N tiles share the A fragment).
// Batch index is wave-uniform (forced scalar via readfirstlane), so all bases
// are SGPRs; loads use SGPR-base + one 32-bit VGPR offset + imm offsets.
// Weights pre-converted to bf16 (w[o][c][t] layout, element stride 9).
// ---------------------------------------------------------------------------
template <bool HASADD>
__global__ __launch_bounds__(128)
void k_conv3x3_wmma(const float* __restrict__ x,
                    const float* __restrict__ xAdd,
                    const __bf16* __restrict__ wb,
                    const float* __restrict__ bias,
                    float* __restrict__ out,
                    int Bn, int Cin, int H, int Wd,
                    int Cout, int outChanOff, int outTotC,
                    int act, const float* __restrict__ scale) {
  const int HWl   = H * Wd;
  const int lane  = threadIdx.x & 31;
  const int wave  = threadIdx.x >> 5;
  const int mo    = blockIdx.x * 16;
  const int pn    = (blockIdx.y * 4 + wave) * 32;   // 32 pixels per wave
  const int total = Bn * HWl;
  if (pn >= total) return;                          // wave-uniform exit

  const int Ktot = Cin * 9;
  const int half = lane >> 4;
  const int l15  = lane & 15;
  const int mrow = min(mo + l15, Cout - 1);         // clamp: dead rows never stored

  // batch index is wave-uniform: HWl % 32 == 0 and pn is a multiple of 32
  const int pns    = __builtin_amdgcn_readfirstlane(pn);
  const int b      = pns / HWl;            // scalar
  const int hwbase = pns - b * HWl;        // scalar
  const int hw0 = hwbase + l15, hw1 = hwbase + 16 + l15;
  const int py0 = hw0 / Wd, px0 = hw0 - py0 * Wd;
  const int py1 = hw1 / Wd, px1 = hw1 - py1 * Wd;

  const float* xb = x + (long)b * Cin * HWl;                      // scalar base
  const float* ab = HASADD ? xAdd + (long)b * Cin * HWl : nullptr;

  v8f acc0 = {}, acc1 = {};
  const int Cmain = Cin & ~31;

  for (int t = 0; t < 9; ++t) {
    const int dy = t / 3 - 1, dx = t - (t / 3) * 3 - 1;
    const int yy0 = py0 + dy, xx0 = px0 + dx;
    const int yy1 = py1 + dy, xx1 = px1 + dx;
    const bool ok0 = ((unsigned)yy0 < (unsigned)H) && ((unsigned)xx0 < (unsigned)Wd);
    const bool ok1 = ((unsigned)yy1 < (unsigned)H) && ((unsigned)xx1 < (unsigned)Wd);
    const int o0 = min(max(yy0, 0), H - 1) * Wd + min(max(xx0, 0), Wd - 1);
    const int o1 = min(max(yy1, 0), H - 1) * Wd + min(max(xx1, 0), Wd - 1);

    // single VGPR offsets; unrolled element offsets become immediates
    int aoff  = mrow * Ktot + t + half * 144;       // elements (stride-9 rows)
    int boff0 = o0 + half * 16 * HWl;               // elements
    int boff1 = o1 + half * 16 * HWl;

    for (int cc = 0; cc < Cmain; cc += 32) {
      v16bf av, bv0, bv1;
#pragma unroll
      for (int e = 0; e < 8; ++e) {
        av[e]     = wb[aoff + e * 9];
        av[e + 8] = wb[aoff + 72 + e * 9];
      }
#pragma unroll
      for (int e = 0; e < 16; ++e) {
        float v0 = xb[boff0 + e * HWl];
        float v1 = xb[boff1 + e * HWl];
        if (HASADD) {
          v0 += ab[boff0 + e * HWl];
          v1 += ab[boff1 + e * HWl];
        }
        bv0[e] = (__bf16)(ok0 ? v0 : 0.0f);
        bv1[e] = (__bf16)(ok1 ? v1 : 0.0f);
      }
      acc0 = __builtin_amdgcn_wmma_f32_16x16x32_bf16(
          false, av, false, bv0, (short)0, acc0, false, false);
      acc1 = __builtin_amdgcn_wmma_f32_16x16x32_bf16(
          false, av, false, bv1, (short)0, acc1, false, false);
      aoff  += 288;        // 32 channels * 9
      boff0 += 32 * HWl;
      boff1 += 32 * HWl;
    }

    if (Cmain < Cin) {  // branchless guarded tail (only conv0: Cin=2)
      const __bf16* wt = wb + mrow * Ktot + t;
      v16bf av, bv0, bv1;
#pragma unroll
      for (int e = 0; e < 16; ++e) {
        int kl = half * 16 + e;
        int c = Cmain + kl;
        __bf16 wv = wt[min(c, Cin - 1) * 9];
        av[e] = (c < Cin) ? wv : (__bf16)0.0f;
      }
#pragma unroll
      for (int e = 0; e < 16; ++e) {
        int c = Cmain + half * 16 + e;
        int cl = min(c, Cin - 1);
        float v0 = xb[cl * HWl + o0];
        float v1 = xb[cl * HWl + o1];
        if (HASADD) {
          v0 += ab[cl * HWl + o0];
          v1 += ab[cl * HWl + o1];
        }
        bv0[e] = (__bf16)((ok0 && c < Cin) ? v0 : 0.0f);
        bv1[e] = (__bf16)((ok1 && c < Cin) ? v1 : 0.0f);
      }
      acc0 = __builtin_amdgcn_wmma_f32_16x16x32_bf16(
          false, av, false, bv0, (short)0, acc0, false, false);
      acc1 = __builtin_amdgcn_wmma_f32_16x16x32_bf16(
          false, av, false, bv1, (short)0, acc1, false, false);
    }
  }

  // C/D layout: N = lane&15, M = r + 8*(lane>=16)
  float* ob = out + (long)b * outTotC * HWl;
  const int obase0 = (outChanOff + mo + 8 * half) * HWl + hw0;
  const int obase1 = (outChanOff + mo + 8 * half) * HWl + hw1;
#pragma unroll
  for (int r = 0; r < 8; ++r) {
    int oc = mo + r + 8 * half;
    if (oc < Cout) {
      float bsv = bias[oc];
      float v0 = apply_act(acc0[r] + bsv, act);
      float v1 = apply_act(acc1[r] + bsv, act);
      if (scale) {
        float sc = scale[b * Cout + oc];
        v0 *= sc;
        v1 *= sc;
      }
      ob[obase0 + r * HWl] = v0;
      ob[obase1 + r * HWl] = v1;
    }
  }
}

// ---------------------------------------------------------------------------
// Modulated deformable conv (DCNv2) as tap-major implicit GEMM. Per tap, the
// offsets/mask/bilinear weights are computed once per pixel; inner loop is
// 4 gathered loads + 3 FMA per B element. Requires Cin % 32 == 0 (512/256/128).
// om: [B,27,H,W] = (o_y[9] | o_x[9] | mask_logits[9])
// ---------------------------------------------------------------------------
__global__ __launch_bounds__(128)
void k_deform_wmma(const float* __restrict__ x,
                   const float* __restrict__ om,
                   const __bf16* __restrict__ wb,
                   const float* __restrict__ bias,
                   float* __restrict__ out,
                   int Bn, int Cin, int H, int Wd,
                   int Cout, int act) {
  const int HWl   = H * Wd;
  const int lane  = threadIdx.x & 31;
  const int wave  = threadIdx.x >> 5;
  const int mo    = blockIdx.x * 16;
  const int pn    = (blockIdx.y * 4 + wave) * 32;
  const int total = Bn * HWl;
  if (pn >= total) return;

  const int Ktot = Cin * 9;
  const int half = lane >> 4;
  const int l15  = lane & 15;
  const int mrow = min(mo + l15, Cout - 1);

  const int pns    = __builtin_amdgcn_readfirstlane(pn);
  const int b      = pns / HWl;
  const int hwbase = pns - b * HWl;
  const int hw0 = hwbase + l15, hw1 = hwbase + 16 + l15;
  const int py0 = hw0 / Wd, px0 = hw0 - py0 * Wd;
  const int py1 = hw1 / Wd, px1 = hw1 - py1 * Wd;

  const float* xb  = x + (long)b * Cin * HWl;     // scalar base
  const float* omB = om + (long)b * 27 * HWl;     // scalar base

  v8f acc0 = {}, acc1 = {};

  for (int t = 0; t < 9; ++t) {
    const int dy = t / 3 - 1, dx = t - (t / 3) * 3 - 1;
    const float* omt = omB + t * HWl;             // scalar base per tap

    int   to[2][4];
    float tw[2][4];
#pragma unroll
    for (int s = 0; s < 2; ++s) {
      int hw = s ? hw1 : hw0;
      int py = s ? py1 : py0;
      int px = s ? px1 : px0;
      float oy = omt[hw];
      float ox = omt[9 * HWl + hw];
      float ml = omt[18 * HWl + hw];
      float m  = 1.0f / (1.0f + __expf(-ml));
      float sy = oy + (float)(dy + py);
      float sx = ox + (float)(dx + px);
      float fy = floorf(sy), fx = floorf(sx);
      float ddy = sy - fy, ddx = sx - fx;
      int iy = (int)fy, ix = (int)fx;
#pragma unroll
      for (int a = 0; a < 2; ++a) {
#pragma unroll
        for (int bq = 0; bq < 2; ++bq) {
          int yq = iy + a, xq = ix + bq;
          bool ok = ((unsigned)yq < (unsigned)H) && ((unsigned)xq < (unsigned)Wd);
          to[s][a * 2 + bq] =
              min(max(yq, 0), H - 1) * Wd + min(max(xq, 0), Wd - 1);
          float wv = (a ? ddy : 1.0f - ddy) * (bq ? ddx : 1.0f - ddx) * m;
          tw[s][a * 2 + bq] = ok ? wv : 0.0f;
        }
      }
    }

    // per-tap VGPR offsets; element offsets folded into immediates
    int aoff = mrow * Ktot + t + half * 144;
    int v00 = to[0][0] + half * 16 * HWl, v01 = to[0][1] + half * 16 * HWl;
    int v02 = to[0][2] + half * 16 * HWl, v03 = to[0][3] + half * 16 * HWl;
    int v10 = to[1][0] + half * 16 * HWl, v11 = to[1][1] + half * 16 * HWl;
    int v12 = to[1][2] + half * 16 * HWl, v13 = to[1][3] + half * 16 * HWl;

    for (int cc = 0; cc < Cin; cc += 32) {
      v16bf av, bv0, bv1;
#pragma unroll
      for (int e = 0; e < 8; ++e) {
        av[e]     = wb[aoff + e * 9];
        av[e + 8] = wb[aoff + 72 + e * 9];
      }
#pragma unroll
      for (int e = 0; e < 16; ++e) {
        float s0 = xb[v00 + e * HWl] * tw[0][0] + xb[v01 + e * HWl] * tw[0][1] +
                   xb[v02 + e * HWl] * tw[0][2] + xb[v03 + e * HWl] * tw[0][3];
        float s1 = xb[v10 + e * HWl] * tw[1][0] + xb[v11 + e * HWl] * tw[1][1] +
                   xb[v12 + e * HWl] * tw[1][2] + xb[v13 + e * HWl] * tw[1][3];
        bv0[e] = (__bf16)s0;
        bv1[e] = (__bf16)s1;
      }
      acc0 = __builtin_amdgcn_wmma_f32_16x16x32_bf16(
          false, av, false, bv0, (short)0, acc0, false, false);
      acc1 = __builtin_amdgcn_wmma_f32_16x16x32_bf16(
          false, av, false, bv1, (short)0, acc1, false, false);
      aoff += 288;
      v00 += 32 * HWl; v01 += 32 * HWl; v02 += 32 * HWl; v03 += 32 * HWl;
      v10 += 32 * HWl; v11 += 32 * HWl; v12 += 32 * HWl; v13 += 32 * HWl;
    }
  }

  float* ob = out + (long)b * Cout * HWl;
  const int obase0 = (mo + 8 * half) * HWl + hw0;
  const int obase1 = (mo + 8 * half) * HWl + hw1;
#pragma unroll
  for (int r = 0; r < 8; ++r) {
    int oc = mo + r + 8 * half;
    if (oc < Cout) {
      float bsv = bias[oc];
      ob[obase0 + r * HWl] = apply_act(acc0[r] + bsv, act);
      ob[obase1 + r * HWl] = apply_act(acc1[r] + bsv, act);
    }
  }
}

// ---------------------------------------------------------------------------
// Elementwise kernels
// ---------------------------------------------------------------------------
__global__ void k_copy_chan(const float* __restrict__ in, long inBstride,
                            float* __restrict__ out, long outBstride,
                            int chOff, int C, int HW, int Bn) {
  long idx = (long)blockIdx.x * blockDim.x + threadIdx.x;
  long total = (long)Bn * C * HW;
  if (idx >= total) return;
  int i = (int)(idx % HW);
  int c = (int)((idx / HW) % C);
  int b = (int)(idx / ((long)C * HW));
  out[(long)b * outBstride + (long)(chOff + c) * HW + i] =
      in[(long)b * inBstride + (long)c * HW + i];
}

__global__ void k_pool_lrelu(const float* __restrict__ in, long inBstride,
                             float* __restrict__ out, int C, int oh, int ow, int Bn) {
  long idx = (long)blockIdx.x * blockDim.x + threadIdx.x;
  long total = (long)Bn * C * oh * ow;
  if (idx >= total) return;
  int xo = (int)(idx % ow); long t = idx / ow;
  int yo = (int)(t % oh); t /= oh;
  int c  = (int)(t % C);
  int b  = (int)(t / C);
  int iw = 2 * ow;
  const float* p = in + (long)b * inBstride + (long)c * (4 * oh * ow);
  float s = 0.25f * (p[(2 * yo) * iw + 2 * xo] + p[(2 * yo) * iw + 2 * xo + 1] +
                     p[(2 * yo + 1) * iw + 2 * xo] + p[(2 * yo + 1) * iw + 2 * xo + 1]);
  out[idx] = s > 0.0f ? s : NEG_SLOPE * s;
}

// align_corners=True bilinear resize into an optional concat channel slice
__global__ void k_up_bilinear(const float* __restrict__ in, int ih, int iw,
                              long inBstride, float* __restrict__ out,
                              int oh, int ow, int outChanOff, int outTotC,
                              int C, int Bn) {
  long idx = (long)blockIdx.x * blockDim.x + threadIdx.x;
  long total = (long)Bn * C * oh * ow;
  if (idx >= total) return;
  int xo = (int)(idx % ow); long t = idx / ow;
  int yo = (int)(t % oh); t /= oh;
  int c  = (int)(t % C);
  int b  = (int)(t / C);
  float ys = (oh > 1) ? (float)yo * (float)(ih - 1) / (float)(oh - 1) : 0.0f;
  float xs = (ow > 1) ? (float)xo * (float)(iw - 1) / (float)(ow - 1) : 0.0f;
  int y0 = min(max((int)floorf(ys), 0), max(ih - 2, 0));
  int x0 = min(max((int)floorf(xs), 0), max(iw - 2, 0));
  float wy = ys - (float)y0, wx = xs - (float)x0;
  int y1 = min(y0 + 1, ih - 1), x1 = min(x0 + 1, iw - 1);
  const float* p = in + (long)b * inBstride + (long)c * ih * iw;
  float v00 = p[(long)y0 * iw + x0], v01 = p[(long)y0 * iw + x1];
  float v10 = p[(long)y1 * iw + x0], v11 = p[(long)y1 * iw + x1];
  float v = (v00 * (1.0f - wx) + v01 * wx) * (1.0f - wy) +
            (v10 * (1.0f - wx) + v11 * wx) * wy;
  out[(long)b * outTotC * oh * ow + (long)(outChanOff + c) * oh * ow +
      (long)yo * ow + xo] = v;
}

__global__ void k_linear_lrelu(const float* __restrict__ in,
                               const float* __restrict__ w,
                               const float* __restrict__ bias,
                               float* __restrict__ out, int Bn, int I, int O) {
  int idx = blockIdx.x * blockDim.x + threadIdx.x;
  if (idx >= Bn * O) return;
  int o = idx % O, b = idx / O;
  const float* ip = in + (long)b * I;
  const float* wp = w + (long)o * I;
  float s = bias[o];
  for (int i = 0; i < I; ++i) s += ip[i] * wp[i];
  out[idx] = s > 0.0f ? s : NEG_SLOPE * s;
}

// ---------------------------------------------------------------------------
// Host orchestration
// ---------------------------------------------------------------------------
extern "C" void kernel_launch(void* const* d_in, const int* in_sizes, int n_in,
                              void* d_out, int out_size, void* d_ws, size_t ws_size,
                              hipStream_t stream) {
  (void)n_in; (void)out_size; (void)ws_size;
  const float* IN  = (const float*)d_in[0];   // [2,1,128,128]
  const float* LAT = (const float*)d_in[1];   // [2,256]
#define P(i) ((const float*)d_in[(i)])

  const int Bn = 2;
  const int H1 = 128, HW1 = 128 * 128;
  const int H2 = 64,  HW2 = 64 * 64;
  const int H4 = 32,  HW4 = 32 * 32;
  const int HW8 = 16 * 16;
  const int H8 = 16;

  // ---- workspace carve-up (floats) ----
  float* ws = (float*)d_ws;
  float* cat0   = ws;                  // [2,2,128,128]
  float* x0     = cat0   + 65536;      // [2,64,128,128]
  float* tA     = x0     + 2097152;    // generic temp (max [2,128,128,128])
  float* tB     = tA     + 4194304;
  float* catD2  = tB     + 4194304;    // [2,256,128,128] = [x1 | up2]
  float* x2b    = catD2  + 8388608;    // [2,128,64,64]
  float* catD1  = x2b    + 1048576;    // [2,512,64,64] = [x3 | up1]
  float* x4b    = catD1  + 4194304;    // [2,256,32,32]
  float* catD0  = x4b    + 524288;     // [2,1024,32,32] = [x5 | up(res4)]
  float* x6b    = catD0  + 2097152;    // [2,512,16,16]
  float* decpre = x6b    + 262144;     // [2,512,16,16]
  float* decin  = decpre + 262144;     // [2,512,32,32]
  float* off1b  = decin  + 1048576;    // [2,128,32,32]
  float* omb    = off1b  + 262144;     // [2,27,128,128] (max)
  float* up1b   = omb    + 884736;     // [2,256,32,32]
  float* up2a   = up1b   + 524288;     // [2,256,64,64]
  float* catO2  = up2a   + 2097152;    // [2,192,64,64] = [o2 | up(off1)]
  float* off2b  = catO2  + 1572864;    // [2,64,64,64]
  float* up2b   = off2b  + 524288;     // [2,128,64,64]
  float* up3a   = up2b   + 1048576;    // [2,128,128,128]
  float* catO3  = up3a   + 4194304;    // [2,96,128,128] = [o3 | up(off2)]
  float* off3b  = catO3  + 3145728;    // [2,32,128,128]
  float* up3b   = off3b  + 1048576;    // [2,64,128,128]
  float* hid    = up3b   + 2097152;    // [2,256]
  float* lp1b   = hid    + 512;        // [2,128]
  float* lp2b   = lp1b   + 256;        // [2,64]
  float* lp3b   = lp2b   + 128;        // [2,32]
  __bf16* warena = (__bf16*)(lp3b + 64);

  // ---- one-time fp32 -> bf16 conversion of all conv/deform weights ----
  static const int WIDX[] = {2, 4, 6, 8, 10, 12, 14, 16, 18, 20, 22, 24, 26,
                             28, 30, 32, 34, 36, 38, 40, 42, 44, 46, 48, 50,
                             52, 54, 56, 58, 60, 62, 64, 66, 68, 70, 72, 74};
  const __bf16* WB[96] = {};
  {
    __bf16* cur = warena;
    for (int j = 0; j < (int)(sizeof(WIDX) / sizeof(WIDX[0])); ++j) {
      int idx = WIDX[j];
      long n = (long)in_sizes[idx];
      k_cvt_bf16<<<dim3((unsigned)((n + 255) / 256)), 256, 0, stream>>>(
          P(idx), cur, n);
      WB[idx] = cur;
      cur += (n + 7) & ~7L;  // keep 16B alignment
    }
  }

  auto conv = [&](const float* x, const float* xadd, int wi, float* out,
                  int Cin, int Hh, int Ww, int Cout, int oco, int otc,
                  int act, const float* scale) {
    dim3 g((Cout + 15) / 16, (Bn * Hh * Ww + 127) / 128);
    if (xadd)
      k_conv3x3_wmma<true><<<g, 128, 0, stream>>>(
          x, xadd, WB[wi], P(wi + 1), out, Bn, Cin, Hh, Ww, Cout, oco, otc,
          act, scale);
    else
      k_conv3x3_wmma<false><<<g, 128, 0, stream>>>(
          x, nullptr, WB[wi], P(wi + 1), out, Bn, Cin, Hh, Ww, Cout, oco, otc,
          act, scale);
  };
  auto deform = [&](const float* x, const float* om, int wi, float* out,
                    int Cin, int Hh, int Ww, int Cout, int act) {
    dim3 g((Cout + 15) / 16, (Bn * Hh * Ww + 127) / 128);
    k_deform_wmma<<<g, 128, 0, stream>>>(x, om, WB[wi], P(wi + 1), out,
                                         Bn, Cin, Hh, Ww, Cout, act);
  };
  auto up = [&](const float* in, int ih, int iw, long inB, float* out,
                int oh, int ow, int oco, int otc, int C) {
    long total = (long)Bn * C * oh * ow;
    k_up_bilinear<<<dim3((unsigned)((total + 255) / 256)), 256, 0, stream>>>(
        in, ih, iw, inB, out, oh, ow, oco, otc, C, Bn);
  };
  auto pool = [&](const float* in, long inB, float* out, int C, int oh, int ow) {
    long total = (long)Bn * C * oh * ow;
    k_pool_lrelu<<<dim3((unsigned)((total + 255) / 256)), 256, 0, stream>>>(
        in, inB, out, C, oh, ow, Bn);
  };
  auto lin = [&](const float* in, int wi, float* out, int I, int O) {
    k_linear_lrelu<<<dim3((Bn * O + 255) / 256), 256, 0, stream>>>(
        in, P(wi), P(wi + 1), out, Bn, I, O);
  };

  // ---- latent MLPs: lp = lrelu(lrelu(x@w1.T+b1)@w2.T+b2) ----
  lin(LAT, 76, hid, 256, 256);  lin(hid, 78, lp1b, 256, 128);
  lin(LAT, 80, hid, 256, 256);  lin(hid, 82, lp2b, 256, 64);
  lin(LAT, 84, hid, 256, 256);  lin(hid, 86, lp3b, 256, 32);

  // ---- stem: x = conv0(concat([input, up(latent 16x16 -> 128x128)])) ----
  {
    long total = (long)Bn * HW1;
    k_copy_chan<<<dim3((unsigned)((total + 255) / 256)), 256, 0, stream>>>(
        IN, HW1, cat0, 2L * HW1, 0, 1, HW1, Bn);
  }
  up(LAT, 16, 16, 256, cat0, 128, 128, 1, 2, 1);
  conv(cat0, nullptr, 2, x0, 2, H1, H1, 64, 0, 64, 0, nullptr);

  // ---- encoder ----
  conv(x0, nullptr, 4, tA, 64, H1, H1, 64, 0, 64, 1, nullptr);          // res1.c1 relu
  conv(tA, nullptr, 6, tB, 64, H1, H1, 64, 0, 64, 0, nullptr);          // res1.c2
  conv(tB, x0, 8, catD2, 64, H1, H1, 128, 0, 256, 0, nullptr);          // x1 -> catD2[:128]
  pool(catD2, 256L * HW1, x2b, 128, H2, H2);                            // x2

  conv(x2b, nullptr, 10, tA, 128, H2, H2, 128, 0, 128, 1, nullptr);     // res2.c1
  conv(tA, nullptr, 12, tB, 128, H2, H2, 128, 0, 128, 0, nullptr);      // res2.c2
  conv(tB, x2b, 14, catD1, 128, H2, H2, 256, 0, 512, 0, nullptr);       // x3 -> catD1[:256]
  pool(catD1, 512L * HW2, x4b, 256, H4, H4);                            // x4

  conv(x4b, nullptr, 16, tA, 256, H4, H4, 256, 0, 256, 1, nullptr);     // res3.c1
  conv(tA, nullptr, 18, tB, 256, H4, H4, 256, 0, 256, 0, nullptr);      // res3.c2
  conv(tB, x4b, 20, catD0, 256, H4, H4, 512, 0, 1024, 0, nullptr);      // x5 -> catD0[:512]
  pool(catD0, 1024L * HW4, x6b, 512, H8, H8);                           // x6

  conv(x6b, nullptr, 22, tA, 512, H8, H8, 512, 0, 512, 1, nullptr);     // res4.c1
  conv(tA, nullptr, 24, tB, 512, H8, H8, 512, 0, 512, 0, nullptr);      // res4.c2
  conv(tB, x6b, 26, decpre, 512, H8, H8, 512, 0, 512, 0, nullptr);      // res4 out

  // ---- decoder level 1 (32x32) ----
  up(decpre, 16, 16, 512L * HW8, catD0, 32, 32, 512, 1024, 512);
  conv(catD0, nullptr, 28, decin, 1024, H4, H4, 512, 0, 512, 0, nullptr);   // dec0
  conv(decin, nullptr, 34, off1b, 512, H4, H4, 128, 0, 128, 2, lp1b);       // off1 = lrelu(c1)*lp1
  conv(off1b, nullptr, 44, omb, 128, H4, H4, 27, 0, 27, 0, nullptr);        // dres1.d1.om
  deform(decin, omb, 46, tA, 512, H4, H4, 512, 2);                          // h = lrelu(dcn1)
  conv(off1b, nullptr, 48, omb, 128, H4, H4, 27, 0, 27, 0, nullptr);        // dres1.d2.om
  deform(tA, omb, 50, tB, 512, H4, H4, 512, 0);                             // dcn2
  conv(tB, decin, 52, up1b, 512, H4, H4, 256, 0, 256, 0, nullptr);          // up1

  // ---- decoder level 2 (64x64) ----
  up(up1b, 32, 32, 256L * HW4, catD1, 64, 64, 256, 512, 256);
  conv(catD1, nullptr, 30, up2a, 512, H2, H2, 256, 0, 256, 0, nullptr);     // dec1
  conv(up2a, nullptr, 36, catO2, 256, H2, H2, 64, 0, 192, 2, nullptr);      // o2 = lrelu(c1)
  up(off1b, 32, 32, 128L * HW4, catO2, 64, 64, 64, 192, 128);               // lo = up(off1)
  conv(catO2, nullptr, 38, off2b, 192, H2, H2, 64, 0, 64, 0, lp2b);         // off2 = c3(cat)*lp2
  conv(off2b, nullptr, 54, omb, 64, H2, H2, 27, 0, 27, 0, nullptr);
  deform(up2a, omb, 56, tA, 256, H2, H2, 256, 2);
  conv(off2b, nullptr, 58, omb, 64, H2, H2, 27, 0, 27, 0, nullptr);
  deform(tA, omb, 60, tB, 256, H2, H2, 256, 0);
  conv(tB, up2a, 62, up2b, 256, H2, H2, 128, 0, 128, 0, nullptr);           // up2

  // ---- decoder level 3 (128x128) ----
  up(up2b, 64, 64, 128L * HW2, catD2, 128, 128, 128, 256, 128);
  conv(catD2, nullptr, 32, up3a, 256, H1, H1, 128, 0, 128, 0, nullptr);     // dec2
  conv(up3a, nullptr, 40, catO3, 128, H1, H1, 32, 0, 96, 2, nullptr);       // o3 = lrelu(c1)
  up(off2b, 64, 64, 64L * HW2, catO3, 128, 128, 32, 96, 64);                // lo = up(off2)
  conv(catO3, nullptr, 42, off3b, 96, H1, H1, 32, 0, 32, 0, lp3b);          // off3
  conv(off3b, nullptr, 64, omb, 32, H1, H1, 27, 0, 27, 0, nullptr);
  deform(up3a, omb, 66, tA, 128, H1, H1, 128, 2);
  conv(off3b, nullptr, 68, omb, 32, H1, H1, 27, 0, 27, 0, nullptr);
  deform(tA, omb, 70, tB, 128, H1, H1, 128, 0);
  conv(tB, up3a, 72, up3b, 128, H1, H1, 64, 0, 64, 0, nullptr);             // up3

  // ---- head ----
  conv(up3b, nullptr, 74, (float*)d_out, 64, H1, H1, 1, 0, 1, 0, nullptr);
#undef P
}
